// GCN_65850438582349
// MI455X (gfx1250) — matrix operand order repaired and checked by reading, not compiled, not run on hardware
//
#include <hip/hip_runtime.h>
#include <hip/hip_bf16.h>

// ---------------------------------------------------------------------------
// Types for CDNA5 WMMA / TDM
// ---------------------------------------------------------------------------
typedef __attribute__((ext_vector_type(16))) __bf16 v16bf;
typedef __attribute__((ext_vector_type(8)))  __bf16 v8bf;
typedef __attribute__((ext_vector_type(8)))  float  v8f;
typedef __attribute__((ext_vector_type(4)))  unsigned int u32x4;
typedef __attribute__((ext_vector_type(8)))  int i32x8;
typedef __attribute__((ext_vector_type(4)))  int i32x4;

#define DEVFN __device__ __forceinline__

DEVFN unsigned short f2bf(float f) {
  unsigned int u = __float_as_uint(f);
  u = (u + 0x7FFFu + ((u >> 16) & 1u)) >> 16;   // round-to-nearest-even
  return (unsigned short)u;
}
DEVFN float bf2f(unsigned short h) {
  return __uint_as_float(((unsigned int)h) << 16);
}

// ---------------------------------------------------------------------------
// TDM: DMA a [rows x 32] bf16 tile (row stride K elems) from global into LDS,
// writing rows of 64B followed by 16B padding (pad_interval=16dw, pad_amount=4dw)
// -> matches LDS layout ushort[128][40]. Rows beyond rowsRem read as zero.
// D# packing per CDNA5 ISA 8.3-8.5. Wave-scalar; issue from one wave only.
// ---------------------------------------------------------------------------
DEVFN void tdm_load_tileA(const unsigned short* gptr, unsigned lds_addr,
                          int rowsRem, int K) {
  unsigned long long ga = (unsigned long long)(size_t)gptr;
  u32x4 g0;
  g0[0] = 1u;                                   // count=1, user mode
  g0[1] = lds_addr;                             // LDS byte address
  g0[2] = (unsigned)ga;                         // global_addr[31:0]
  g0[3] = (unsigned)((ga >> 32) & 0x01FFFFFFu)  // global_addr[56:32]
        | (2u << 30);                           // type=2 (image)
  unsigned rows = (unsigned)rowsRem;
  i32x8 g1;
  g1[0] = (int)((1u << 16)                      // data_size=1 -> 2 bytes
              | (1u << 20)                      // pad_enable
              | (3u << 22)                      // pad_interval: 16 DWORDs (64B)
              | (3u << 25));                    // pad_amount: 4 DWORDs (16B)
  g1[1] = (int)(((unsigned)K & 0xFFFFu) << 16); // tensor_dim0[15:0]
  g1[2] = (int)((rows & 0xFFFFu) << 16);        // dim0[31:16]=0 | tensor_dim1[15:0]
  g1[3] = (int)((rows >> 16) | (32u << 16));    // tensor_dim1[31:16] | tile_dim0=32
  g1[4] = 128;                                  // tile_dim1=128, tile_dim2=0
  g1[5] = K;                                    // tensor_dim0_stride[31:0]
  g1[6] = 0;
  g1[7] = 0;
  i32x4 z4 = {0, 0, 0, 0};                      // 2D tensor: groups 2/3 unused
  i32x8 z8 = {0, 0, 0, 0, 0, 0, 0, 0};
  __builtin_amdgcn_tensor_load_to_lds(g0, g1, z4, z4, z8, 0);
}

// ---------------------------------------------------------------------------
// Small elementwise kernels
// ---------------------------------------------------------------------------
__global__ void gcn_fill_f32(float* __restrict__ p, float v, long long n) {
  long long i = (long long)blockIdx.x * blockDim.x + threadIdx.x;
  if (i < n) p[i] = v;
}

__global__ void gcn_f32_to_bf16(const float* __restrict__ s,
                                unsigned short* __restrict__ d, long long n) {
  long long i = (long long)blockIdx.x * blockDim.x + threadIdx.x;
  if (i < n) d[i] = f2bf(s[i]);
}

__global__ void gcn_deg_accum(const int* __restrict__ dst, float* __restrict__ deg, int E) {
  int e = blockIdx.x * blockDim.x + threadIdx.x;
  if (e < E) atomicAdd(&deg[dst[e]], 1.0f);
}

__global__ void gcn_deg_finalize(const float* __restrict__ deg,
                                 float* __restrict__ dinv,
                                 float* __restrict__ selfn, int N) {
  int i = blockIdx.x * blockDim.x + threadIdx.x;
  if (i < N) {
    float d = deg[i];
    dinv[i]  = rsqrtf(d);
    selfn[i] = 1.0f / d;
  }
}

// agg[dst] += hW[src] * (dinv[src]*dinv[dst]); float4-vectorized, f fast for coalescing
__global__ void gcn_edge_aggregate(const float* __restrict__ hW,
                                   const int* __restrict__ src,
                                   const int* __restrict__ dst,
                                   const float* __restrict__ dinv,
                                   float* __restrict__ agg, int E, int H) {
  long long idx = (long long)blockIdx.x * blockDim.x + threadIdx.x;
  int chunks = H >> 2;
  long long e = idx / chunks;
  if (e >= E) return;
  int f = (int)(idx % chunks) << 2;
  int s = src[e], d = dst[e];
  float en = dinv[s] * dinv[d];
  const float4 v = *(const float4*)&hW[(size_t)s * H + f];
  float* ap = &agg[(size_t)d * H + f];
  atomicAdd(ap + 0, v.x * en);
  atomicAdd(ap + 1, v.y * en);
  atomicAdd(ap + 2, v.z * en);
  atomicAdd(ap + 3, v.w * en);
}

// ---------------------------------------------------------------------------
// Fused conv finalize: t = agg + hW*selfnorm + bias ; LayerNorm(t) ; relu? ;
// + addend ; write fp32 and bf16.   One block (128 threads) per node row.
// ---------------------------------------------------------------------------
__global__ __launch_bounds__(128)
void gcn_conv_finalize_ln(const float* __restrict__ agg,
                          const float* __restrict__ hW,
                          const float* __restrict__ selfn,
                          const float* __restrict__ bias,
                          const float* __restrict__ g,
                          const float* __restrict__ bvec,
                          const float* __restrict__ addend,
                          float* __restrict__ outf,
                          unsigned short* __restrict__ outbf,
                          int H, int relu) {
  __shared__ float tv[512];
  __shared__ float red[128];
  const int t = threadIdx.x;
  const int row = blockIdx.x;
  const size_t base = (size_t)row * H;
  const float sn = selfn[row];
  const float invH = 1.0f / (float)H;

  float s = 0.0f;
  for (int f = t; f < H; f += 128) {
    float v = agg[base + f] + hW[base + f] * sn + bias[f];
    tv[f] = v;
    s += v;
  }
  red[t] = s; __syncthreads();
  for (int o = 64; o > 0; o >>= 1) { if (t < o) red[t] += red[t + o]; __syncthreads(); }
  const float mu = red[0] * invH;
  __syncthreads();

  float s2 = 0.0f;
  for (int f = t; f < H; f += 128) { float d = tv[f] - mu; s2 += d * d; }
  red[t] = s2; __syncthreads();
  for (int o = 64; o > 0; o >>= 1) { if (t < o) red[t] += red[t + o]; __syncthreads(); }
  const float rs = rsqrtf(red[0] * invH + 1e-5f);
  __syncthreads();

  for (int f = t; f < H; f += 128) {
    float y = (tv[f] - mu) * rs * g[f] + bvec[f];
    if (relu) y = fmaxf(y, 0.0f);
    if (addend) y += addend[base + f];
    if (outf)  outf[base + f]  = y;
    if (outbf) outbf[base + f] = f2bf(y);
  }
}

// ---------------------------------------------------------------------------
// Gather-concat eval edge features: e[i] = [x4[s_i], x4[d_i]]  (bf16 out)
// ---------------------------------------------------------------------------
__global__ void gcn_gather_concat(const float* __restrict__ x4,
                                  const int* __restrict__ s,
                                  const int* __restrict__ d,
                                  unsigned short* __restrict__ out, int EV) {
  long long idx = (long long)blockIdx.x * blockDim.x + threadIdx.x;
  long long total = (long long)EV * 512;
  if (idx >= total) return;
  int i = (int)(idx >> 9);
  int f = (int)(idx & 511);
  int node = (f < 256) ? s[i] : d[i];
  out[idx] = f2bf(x4[(size_t)node * 256 + (f & 255)]);
}

// ---------------------------------------------------------------------------
// Final 128->1 dot + sigmoid, one thread per eval edge
// ---------------------------------------------------------------------------
__global__ void gcn_lp_final(const unsigned short* __restrict__ h,
                             const float* __restrict__ w,
                             const float* __restrict__ b,
                             float* __restrict__ out, int M, int K) {
  int i = blockIdx.x * blockDim.x + threadIdx.x;
  if (i >= M) return;
  float acc = b[0];
  const unsigned short* hp = h + (size_t)i * K;
  for (int f = 0; f < K; ++f) acc += bf2f(hp[f]) * w[f];
  out[i] = 1.0f / (1.0f + __expf(-acc));
}

// ---------------------------------------------------------------------------
// bf16 WMMA GEMM: C[M,Nc] = A[M,K] @ B[K,Nc]  (+epilogue)
// Block: 256 threads = 8 waves, tile 128x64 (BK=32). Each wave: 32x32 via
// 2x2 v_wmma_f32_16x16x32_bf16. A tiles staged by the Tensor Data Mover
// (double-buffered, s_wait_tensorcnt); B staged transposed by VALU so
// fragments are contiguous ds_load_b128s.
// mode: 0=none 1=+bias 2=+bias,relu 3=+bias,relu,BN
// ---------------------------------------------------------------------------
__global__ __launch_bounds__(256, 2)
void gcn_wmma_gemm(const unsigned short* __restrict__ A,
                   const unsigned short* __restrict__ B,
                   float* __restrict__ Cf,
                   unsigned short* __restrict__ Cbf,
                   const float* __restrict__ bias,
                   const float* __restrict__ bng, const float* __restrict__ bnb,
                   const float* __restrict__ bnm, const float* __restrict__ bnv,
                   int M, int K, int Nc, int mode) {
  __shared__ unsigned short As[2][128][40];   // 64B rows + 16B pad (TDM pad feature)
  __shared__ unsigned short Bs[2][64][40];    // transposed: Bs[buf][n][k]

  const int tid  = threadIdx.x;
  const int lane = tid & 31;
  const int wave = tid >> 5;
  const int wm = wave & 3;        // 4 waves down M
  const int wn = wave >> 2;       // 2 waves across N
  const int half = lane >> 4;     // half-wave (VGPR layout split)
  const int l16  = lane & 15;
  const int blockM = blockIdx.x * 128;
  const int blockN = blockIdx.y * 64;
  const int rowsRem = M - blockM;

  v8f zero;
  #pragma unroll
  for (int i = 0; i < 8; ++i) zero[i] = 0.0f;
  v8f acc[2][2];
  acc[0][0] = zero; acc[0][1] = zero; acc[1][0] = zero; acc[1][1] = zero;

  const int akb = half ? 8 : 0;    // A: lanes 0-15 -> K {0..7,16..23}; 16-31 -> {8..15,24..31}
  const int bkb = half ? 8 : 0;    // B: lanes 0-15 -> K {0..7,16..23}; 16-31 -> {8..15,24..31}

  const unsigned ldsA0 = (unsigned)(size_t)&As[0][0][0];
  const unsigned ldsA1 = (unsigned)(size_t)&As[1][0][0];
  const unsigned short* Atile = A + (size_t)blockM * K;

  // Prologue: kick off TDM for the first A tile
  if (wave == 0) tdm_load_tileA(Atile, ldsA0, rowsRem, K);

  int buf = 0;
  for (int k0 = 0; k0 < K; k0 += 32) {
    // Stage B tile 32x64 transposed -> Bs[buf][n][k] (coalesced global reads over n)
    for (int i = tid; i < 64 * 32; i += 256) {
      int k = i >> 6, n = i & 63;
      Bs[buf][n][k] = B[(size_t)(k0 + k) * Nc + blockN + n];
    }
    if (wave == 0) __builtin_amdgcn_s_wait_tensorcnt(0);   // A[buf] DMA complete
    __syncthreads();                                       // publish A[buf], B[buf]
    // Overlap: start DMA for the next A tile while we compute on this one
    if (wave == 0 && (k0 + 32) < K)
      tdm_load_tileA(Atile + (k0 + 32), (buf ? ldsA0 : ldsA1), rowsRem, K);

    // Build fragments per ISA VGPR layout and issue 2x2 WMMAs
    v16bf afr[2], bfr[2];
    #pragma unroll
    for (int t = 0; t < 2; ++t) {
      int r = wm * 32 + t * 16 + l16;
      v16bf a;
      ((v8bf*)&a)[0] = *(const v8bf*)&As[buf][r][akb];
      ((v8bf*)&a)[1] = *(const v8bf*)&As[buf][r][akb + 16];
      afr[t] = a;
      int c = wn * 32 + t * 16 + l16;
      v16bf b;
      ((v8bf*)&b)[0] = *(const v8bf*)&Bs[buf][c][bkb];
      ((v8bf*)&b)[1] = *(const v8bf*)&Bs[buf][c][bkb + 16];
      bfr[t] = b;
    }
    #pragma unroll
    for (int tm = 0; tm < 2; ++tm)
      #pragma unroll
      for (int tn = 0; tn < 2; ++tn)
        acc[tm][tn] = __builtin_amdgcn_wmma_f32_16x16x32_bf16(
            false, afr[tm], false, bfr[tn], (short)0, acc[tm][tn], false, false);

    buf ^= 1;
    __syncthreads();   // compute done before next iteration's staging overwrites
  }

  // Epilogue + store (C layout: VGPR j -> row j (+8 upper half); col = l16)
  #pragma unroll
  for (int tm = 0; tm < 2; ++tm) {
    #pragma unroll
    for (int tn = 0; tn < 2; ++tn) {
      int col = blockN + wn * 32 + tn * 16 + l16;
      float bia = (mode >= 1) ? bias[col] : 0.0f;
      float sc = 1.0f, mm = 0.0f, bb = 0.0f;
      if (mode == 3) {
        sc = bng[col] * rsqrtf(bnv[col] + 1e-5f);
        mm = bnm[col]; bb = bnb[col];
      }
      #pragma unroll
      for (int j = 0; j < 8; ++j) {
        int row = blockM + wm * 32 + tm * 16 + j + (half ? 8 : 0);
        if (row < M) {
          float y = acc[tm][tn][j] + bia;
          if (mode >= 2) y = fmaxf(y, 0.0f);
          if (mode == 3) y = (y - mm) * sc + bb;
          size_t o = (size_t)row * Nc + col;
          if (Cf)  Cf[o]  = y;
          if (Cbf) Cbf[o] = f2bf(y);
        }
      }
    }
  }
}

// ---------------------------------------------------------------------------
// Host orchestration
// ---------------------------------------------------------------------------
extern "C" void kernel_launch(void* const* d_in, const int* in_sizes, int n_in,
                              void* d_out, int out_size, void* d_ws, size_t ws_size,
                              hipStream_t stream) {
  (void)in_sizes; (void)n_in; (void)out_size; (void)ws_size;
  const int N = 100000, E = 800000, EV = 200000;
  const int IN = 128, H = 512, OUT = 256;

  // Inputs in setup_inputs() dict order
  const float* x       = (const float*)d_in[0];
  const int*   ei      = (const int*)d_in[1];
  const int*   eval_ei = (const int*)d_in[2];
  const float* W1 = (const float*)d_in[3];  const float* b1 = (const float*)d_in[4];
  const float* W2 = (const float*)d_in[5];  const float* b2 = (const float*)d_in[6];
  const float* W3 = (const float*)d_in[7];  const float* b3 = (const float*)d_in[8];
  const float* W4 = (const float*)d_in[9];  const float* b4 = (const float*)d_in[10];
  const float* ln1g = (const float*)d_in[11]; const float* ln1b = (const float*)d_in[12];
  const float* ln2g = (const float*)d_in[13]; const float* ln2b = (const float*)d_in[14];
  const float* ln3g = (const float*)d_in[15]; const float* ln3b = (const float*)d_in[16];
  const float* ln4g = (const float*)d_in[17]; const float* ln4b = (const float*)d_in[18];
  const float* r1W = (const float*)d_in[19]; const float* r1b = (const float*)d_in[20];
  const float* r3W = (const float*)d_in[21]; const float* r3b = (const float*)d_in[22];
  const float* l1W = (const float*)d_in[23]; const float* l1b = (const float*)d_in[24];
  const float* l2W = (const float*)d_in[25]; const float* l2b = (const float*)d_in[26];
  const float* l3W = (const float*)d_in[27]; const float* l3b = (const float*)d_in[28];
  const float* l4W = (const float*)d_in[29]; const float* l4b = (const float*)d_in[30];
  const float* bn1g = (const float*)d_in[31]; const float* bn1b = (const float*)d_in[32];
  const float* bn1m = (const float*)d_in[33]; const float* bn1v = (const float*)d_in[34];
  const float* bn2g = (const float*)d_in[35]; const float* bn2b = (const float*)d_in[36];
  const float* bn2m = (const float*)d_in[37]; const float* bn2v = (const float*)d_in[38];
  const float* bn3g = (const float*)d_in[39]; const float* bn3b = (const float*)d_in[40];
  const float* bn3m = (const float*)d_in[41]; const float* bn3v = (const float*)d_in[42];

  const int* src = ei;          const int* dstv = ei + E;
  const int* es  = eval_ei;     const int* ed   = eval_ei + EV;

  // Workspace carve (aligned to 256B)
  char* p = (char*)d_ws;
  auto carve = [&](size_t bytes) -> char* {
    char* r = p; p += (bytes + 255) & ~(size_t)255; return r;
  };
  float* deg   = (float*)carve((size_t)N * 4);
  float* dinv  = (float*)carve((size_t)N * 4);
  float* selfn = (float*)carve((size_t)N * 4);
  float* hW    = (float*)carve((size_t)N * H * 4);   // reused as bufP (EV*512 bf16)
  float* agg   = (float*)carve((size_t)N * H * 4);   // reused as bufQ
  float* resf  = (float*)carve((size_t)N * H * 4);
  float* x1f   = (float*)carve((size_t)N * H * 4);
  float* xcurf = (float*)carve((size_t)N * H * 4);
  float* x4f   = (float*)carve((size_t)N * OUT * 4);
  unsigned short* xinbf  = (unsigned short*)carve((size_t)N * IN * 2);
  unsigned short* x1bf   = (unsigned short*)carve((size_t)N * H * 2);
  unsigned short* xcurbf = (unsigned short*)carve((size_t)N * H * 2);
  unsigned short* wb     = (unsigned short*)carve((size_t)1343488 * 2);
  unsigned short* W1b = wb;            unsigned short* W2b = W1b + 65536;
  unsigned short* W3b = W2b + 262144;  unsigned short* W4b = W3b + 262144;
  unsigned short* R1b = W4b + 131072;  unsigned short* R3b = R1b + 65536;
  unsigned short* L1b = R3b + 131072;  unsigned short* L2b = L1b + 262144;
  unsigned short* L3b = L2b + 131072;
  unsigned short* bufP = (unsigned short*)hW;    // EV*512 bf16 = N*H*4 bytes exactly
  unsigned short* bufQ = (unsigned short*)agg;

  auto cvt = [&](const float* s, unsigned short* d, long long n) {
    gcn_f32_to_bf16<<<(unsigned)((n + 255) / 256), 256, 0, stream>>>(s, d, n);
  };
  auto fill = [&](float* q, float v, long long n) {
    gcn_fill_f32<<<(unsigned)((n + 255) / 256), 256, 0, stream>>>(q, v, n);
  };
  auto gemm = [&](const unsigned short* A, const unsigned short* B, float* Cf,
                  unsigned short* Cbf, const float* bias,
                  const float* g, const float* bb, const float* m, const float* v,
                  int M, int K, int Nc, int mode) {
    dim3 grid((M + 127) / 128, Nc / 64);
    gcn_wmma_gemm<<<grid, 256, 0, stream>>>(A, B, Cf, Cbf, bias, g, bb, m, v, M, K, Nc, mode);
  };
  auto aggregate = [&](const float* hw, int Hc) {
    long long tot = (long long)E * (Hc >> 2);
    gcn_edge_aggregate<<<(unsigned)((tot + 255) / 256), 256, 0, stream>>>(
        hw, src, dstv, dinv, agg, E, Hc);
  };
  auto finalize = [&](const float* bias, const float* g, const float* bv,
                      const float* addend, float* of, unsigned short* ob,
                      int Hc, int relu) {
    gcn_conv_finalize_ln<<<N, 128, 0, stream>>>(agg, hW, selfn, bias, g, bv,
                                                addend, of, ob, Hc, relu);
  };

  // 1) bf16 weight/input conversions
  cvt(W1, W1b, 65536);   cvt(W2, W2b, 262144); cvt(W3, W3b, 262144);
  cvt(W4, W4b, 131072);  cvt(r1W, R1b, 65536); cvt(r3W, R3b, 131072);
  cvt(l1W, L1b, 262144); cvt(l2W, L2b, 131072); cvt(l3W, L3b, 32768);
  cvt(x, xinbf, (long long)N * IN);

  // 2) degrees / norms
  fill(deg, 1.0f, N);
  gcn_deg_accum<<<(E + 255) / 256, 256, 0, stream>>>(dstv, deg, E);
  gcn_deg_finalize<<<(N + 255) / 256, 256, 0, stream>>>(deg, dinv, selfn, N);

  // 3) conv1 + res1 + LN/relu  -> x1
  gemm(xinbf, W1b, hW, nullptr, nullptr, nullptr, nullptr, nullptr, nullptr, N, IN, H, 0);
  fill(agg, 0.0f, (long long)N * H);
  aggregate(hW, H);
  gemm(xinbf, R1b, resf, nullptr, r1b, nullptr, nullptr, nullptr, nullptr, N, IN, H, 1);
  finalize(b1, ln1g, ln1b, resf, x1f, x1bf, H, 1);

  // 4) conv2 -> x2
  gemm(x1bf, W2b, hW, nullptr, nullptr, nullptr, nullptr, nullptr, nullptr, N, H, H, 0);
  fill(agg, 0.0f, (long long)N * H);
  aggregate(hW, H);
  finalize(b2, ln2g, ln2b, x1f, xcurf, xcurbf, H, 1);

  // 5) conv3 -> x3 (in-place over x2 buffers; per-element read-then-write is safe)
  gemm(xcurbf, W3b, hW, nullptr, nullptr, nullptr, nullptr, nullptr, nullptr, N, H, H, 0);
  fill(agg, 0.0f, (long long)N * H);
  aggregate(hW, H);
  finalize(b3, ln3g, ln3b, xcurf, xcurf, xcurbf, H, 1);

  // 6) res3 + conv4 -> x4 (no relu)
  gemm(x1bf, R3b, resf, nullptr, r3b, nullptr, nullptr, nullptr, nullptr, N, H, OUT, 1);
  gemm(xcurbf, W4b, hW, nullptr, nullptr, nullptr, nullptr, nullptr, nullptr, N, H, OUT, 0);
  fill(agg, 0.0f, (long long)N * OUT);
  aggregate(hW, OUT);
  finalize(b4, ln4g, ln4b, resf, x4f, nullptr, OUT, 0);

  // 7) link-prediction head (hW/agg regions reused as bf16 ping-pong buffers)
  {
    long long tot = (long long)EV * 512;
    gcn_gather_concat<<<(unsigned)((tot + 255) / 256), 256, 0, stream>>>(x4f, es, ed, bufP, EV);
  }
  gemm(bufP, L1b, nullptr, bufQ, l1b, bn1g, bn1b, bn1m, bn1v, EV, 512, 512, 3);
  gemm(bufQ, L2b, nullptr, bufP, l2b, bn2g, bn2b, bn2m, bn2v, EV, 512, 256, 3);
  gemm(bufP, L3b, nullptr, bufQ, l3b, bn3g, bn3b, bn3m, bn3v, EV, 256, 128, 3);
  gcn_lp_final<<<(EV + 255) / 256, 256, 0, stream>>>(bufQ, l4W, l4b, (float*)d_out, EV, 128);
}
